// RouteNetGauss_38809324487239
// MI455X (gfx1250) — matrix operand order repaired and checked by reading, not compiled, not run on hardware
//
// RouteNet-Gauss forward for MI455X (gfx1250), wave32 + WMMA f16->f32.
// v2: GEMM templated on KTILES (removes per-tile scalar branches) and all B tiles
// preloaded into registers before the WMMA sequence (loads issue as one clause,
// WMMAs drain back-to-back instead of each waiting on s_wait_loadcnt 0).
#include <hip/hip_runtime.h>
#include <cstdint>
#include <cstddef>

#define NF 40000
#define NP 6
#define NS 2
#define NL 8000
#define NQ 8000
#define NM 30
#define NN 2000
#define NG 4
#define NH 32
#define NITERS 4

typedef _Float16 f16;
typedef __attribute__((ext_vector_type(16))) _Float16 v16h;
typedef __attribute__((ext_vector_type(8))) float v8f;

__device__ __forceinline__ float sigmf_(float x) { return 1.0f / (1.0f + expf(-x)); }
__device__ __forceinline__ float reluf_(float x) { return x > 0.0f ? x : 0.0f; }

// ---- A tile loader: 16x32 f32 row-major -> WMMA 16-bit A layout (wave32) ----
// lanes 0-15: M=lane, K elements {0..7, 16..23}; lanes 16-31: M=lane-16, K {8..15, 24..31}.
// 'ar' = row base + kb (kb = 0 for lanes<16, 8 for lanes>=16).
__device__ __forceinline__ v16h load_a16x32(const float* __restrict__ ar) {
  float4 x0 = *(const float4*)(ar + 0);
  float4 x1 = *(const float4*)(ar + 4);
  float4 x2 = *(const float4*)(ar + 16);
  float4 x3 = *(const float4*)(ar + 20);
  v16h a;
  a[0] = (f16)x0.x;  a[1] = (f16)x0.y;  a[2]  = (f16)x0.z;  a[3]  = (f16)x0.w;
  a[4] = (f16)x1.x;  a[5] = (f16)x1.y;  a[6]  = (f16)x1.z;  a[7]  = (f16)x1.w;
  a[8] = (f16)x2.x;  a[9] = (f16)x2.y;  a[10] = (f16)x2.z;  a[11] = (f16)x2.w;
  a[12] = (f16)x3.x; a[13] = (f16)x3.y; a[14] = (f16)x3.z;  a[15] = (f16)x3.w;
  return a;
}

// ---- B tile loader: prepacked [ktile][ntile][lane][8 dwords] (lane&15 = col n, dword d = K pair, A's k order)
__device__ __forceinline__ v16h load_b_tile(const unsigned int* __restrict__ Bp, int kt, int j, int lane) {
  const uint4* B4 = (const uint4*)Bp;
  int base = ((kt * 6 + j) * 32 + lane) * 2;
  union { uint4 q[2]; v16h h; } b;
  b.q[0] = B4[base];
  b.q[1] = B4[base + 1];
  return b.h;
}

// ---- pack weight [K x 96] fp32 row-major into f16 B layout ----
__global__ void pack_b_kernel(const float* __restrict__ W, unsigned int* __restrict__ Bp, int K) {
  int t = blockIdx.x * blockDim.x + threadIdx.x;
  int total = (K >> 5) * 6 * 32 * 8;
  if (t >= total) return;
  int d = t & 7;
  int lane = (t >> 3) & 31;
  int rest = t >> 8;
  int j = rest % 6;
  int kt = rest / 6;
  int k = kt * 32 + 8 * (lane >> 4) + 2 * d + ((d >= 4) ? 8 : 0);
  int n = j * 16 + (lane & 15);
  union { f16 h[2]; unsigned int u; } pk;
  pk.h[0] = (f16)W[k * 96 + n];
  pk.h[1] = (f16)W[(k + 1) * 96 + n];
  Bp[t] = pk.u;
}

// ---- generic GEMM: Out[M x 96] = concat_A @ Wpacked + bias ----
// kt=0 rows from A0 (optional gather idx0), kt=1 rows from A1 (optional gather idx1).
// Rows of A0/A1 are 32 floats. M must be a multiple of 16.
template <int KTILES>
__global__ void __launch_bounds__(256)
gemm96_kernel(const float* __restrict__ A0, const float* __restrict__ A1,
              const int* __restrict__ idx0, const int* __restrict__ idx1, int idxStride,
              const unsigned int* __restrict__ Bp, const float* __restrict__ bias,
              float* __restrict__ Out, int M) {
  int lane = threadIdx.x & 31;
  int wave = threadIdx.x >> 5;
  int row0 = (blockIdx.x * 8 + wave) * 16;
  if (row0 >= M) return;                       // wave-uniform; EXEC stays full for active waves
  int mrow = row0 + (lane & 15);
  int kb = (lane & 16) ? 8 : 0;

  int r0i = idx0 ? idx0[(size_t)mrow * idxStride] : mrow;
  v16h a0 = load_a16x32(A0 + (size_t)r0i * NH + kb);
  v16h a1 = a0;
  if (KTILES == 2) {
    int r1i = idx1 ? idx1[(size_t)mrow * idxStride] : mrow;
    a1 = load_a16x32(A1 + (size_t)r1i * NH + kb);
  }

  // preload all B tiles so the WMMA sequence is not gated load-by-load
  v16h b[KTILES * 6];
#pragma unroll
  for (int kt = 0; kt < KTILES; ++kt)
#pragma unroll
    for (int j = 0; j < 6; ++j) b[kt * 6 + j] = load_b_tile(Bp, kt, j, lane);

  v8f acc[6];
#pragma unroll
  for (int j = 0; j < 6; ++j) {
    v8f z = {};
    z = __builtin_amdgcn_wmma_f32_16x16x32_f16(false, a0, false, b[j], (short)0, z, false, false);
    if (KTILES == 2)
      z = __builtin_amdgcn_wmma_f32_16x16x32_f16(false, a1, false, b[6 + j], (short)0, z, false, false);
    acc[j] = z;
  }

  int c = lane & 15;
  int mb = row0 + ((lane & 16) ? 8 : 0);
#pragma unroll
  for (int j = 0; j < 6; ++j) {
    int n = j * 16 + c;
    float bv = bias[n];
#pragma unroll
    for (int r8 = 0; r8 < 8; ++r8)
      Out[(size_t)(mb + r8) * 96 + n] = acc[j][r8] + bv;  // lanes of one VGPR are contiguous in n
  }
}

// ---- fused gh = H@Wh (+bh) and GRU gates; H updated in place; optional seq_full write ----
// C-layout epilogue: lane holds (z,r,n) for hidden cols c and c+16 of 8 rows.
__global__ void __launch_bounds__(256)
gru_hgate_kernel(float* __restrict__ H, const unsigned int* __restrict__ BpWh,
                 const float* __restrict__ bh,
                 const float* __restrict__ gx, int gxStride, int gxOff,
                 float* __restrict__ seq_out, int seqStride, int M) {
  int lane = threadIdx.x & 31;
  int wave = threadIdx.x >> 5;
  int row0 = (blockIdx.x * 8 + wave) * 16;
  if (row0 >= M) return;
  int mrow = row0 + (lane & 15);
  int kb = (lane & 16) ? 8 : 0;
  v16h a = load_a16x32(H + (size_t)mrow * NH + kb);     // A consumed before H is overwritten

  v16h b[6];
#pragma unroll
  for (int j = 0; j < 6; ++j) b[j] = load_b_tile(BpWh, 0, j, lane);

  v8f acc[6];
#pragma unroll
  for (int j = 0; j < 6; ++j) {
    v8f z = {};
    acc[j] = __builtin_amdgcn_wmma_f32_16x16x32_f16(false, a, false, b[j], (short)0, z, false, false);
  }
  int c = lane & 15;
  int mb = row0 + ((lane & 16) ? 8 : 0);
  float bz0 = bh[c], bz1 = bh[16 + c];
  float br0 = bh[32 + c], br1 = bh[48 + c];
  float bn0 = bh[64 + c], bn1 = bh[80 + c];
#pragma unroll
  for (int r8 = 0; r8 < 8; ++r8) {
    int m = mb + r8;
    size_t gr = ((size_t)m * gxStride + gxOff) * 96;
    float z0 = sigmf_(gx[gr + c]      + acc[0][r8] + bz0);
    float z1 = sigmf_(gx[gr + 16 + c] + acc[1][r8] + bz1);
    float rr0 = sigmf_(gx[gr + 32 + c] + acc[2][r8] + br0);
    float rr1 = sigmf_(gx[gr + 48 + c] + acc[3][r8] + br1);
    float n0 = tanhf(gx[gr + 64 + c] + rr0 * (acc[4][r8] + bn0));
    float n1 = tanhf(gx[gr + 80 + c] + rr1 * (acc[5][r8] + bn1));
    size_t hb = (size_t)m * NH;
    float o0 = z0 * H[hb + c]      + (1.0f - z0) * n0;
    float o1 = z1 * H[hb + 16 + c] + (1.0f - z1) * n1;
    H[hb + c] = o0;
    H[hb + 16 + c] = o1;
    if (seq_out) {
      seq_out[(size_t)m * seqStride + c] = o0;
      seq_out[(size_t)m * seqStride + 16 + c] = o1;
    }
  }
}

// ---- seq_full[:,0,:] = h ----
__global__ void seq0_copy_kernel(const float* __restrict__ h, float* __restrict__ seqf) {
  int t = blockIdx.x * blockDim.x + threadIdx.x;
  if (t >= NF * NH) return;
  int m = t >> 5, j = t & 31;
  seqf[(size_t)m * ((NP + 1) * NH) + j] = h[t];
}

// ---- flow_sum[q] = sum_m seq_full[fid[q,m], pos[q,m]] ----
__global__ void flow_sum_kernel(const int* __restrict__ ftq, const float* __restrict__ seqf,
                                float* __restrict__ fsum) {
  int t = blockIdx.x * blockDim.x + threadIdx.x;
  if (t >= NQ * NH) return;
  int q = t >> 5, j = t & 31;
  float s = 0.0f;
  for (int m = 0; m < NM; ++m) {
    int base = (q * NM + m) * 2;
    int fid = ftq[base];
    int pos = ftq[base + 1];
    s += seqf[((size_t)fid * (NP + 1) + pos) * NH + j];
  }
  fsum[t] = s;
}

// ---- xn[n] = sum_g queue_state[node_groupings[n,g]] ----
__global__ void node_sum_kernel(const int* __restrict__ grp, const float* __restrict__ qs,
                                float* __restrict__ xn) {
  int t = blockIdx.x * blockDim.x + threadIdx.x;
  if (t >= NN * NH) return;
  int n = t >> 5, j = t & 31;
  float s = 0.0f;
  for (int g = 0; g < NG; ++g) s += qs[(size_t)grp[n * NG + g] * NH + j];
  xn[t] = s;
}

// ---- flow embedding: feats(3) -> 32 -> 32, relu; flow_has_traffic is all-ones by construction ----
__global__ void flow_emb_kernel(const float* __restrict__ traffic, const float* __restrict__ rate,
                                const float* __restrict__ W1, const float* __restrict__ b1,
                                const float* __restrict__ W2, const float* __restrict__ b2,
                                float* __restrict__ init_flow) {
  int t = blockIdx.x * blockDim.x + threadIdx.x;
  if (t >= NF * NS) return;
  int f = t / NS, s = t % NS;
  float x0 = (traffic[t] - 500.0f) * (1.0f / 250.0f);
  float x1 = (rate[t] - 1.0f) * 2.0f;
  float x2 = 1.0f;
  float h1[NH];
  for (int j = 0; j < NH; ++j)
    h1[j] = reluf_(x0 * W1[j] + x1 * W1[NH + j] + x2 * W1[2 * NH + j] + b1[j]);
  float* outp = init_flow + ((size_t)s * NF + f) * NH;
  for (int j = 0; j < NH; ++j) {
    float acc = b2[j];
    for (int i = 0; i < NH; ++i) acc += h1[i] * W2[i * NH + j];
    outp[j] = reluf_(acc);
  }
}

// ---- link embedding incl. traffic aggregation: load scalar -> 32 -> 32, relu ----
__global__ void link_emb_kernel(const int* __restrict__ ftq, const float* __restrict__ traffic,
                                const float* __restrict__ rate, const float* __restrict__ hdr,
                                const float* __restrict__ cap,
                                const float* __restrict__ W1, const float* __restrict__ b1,
                                const float* __restrict__ W2, const float* __restrict__ b2,
                                float* __restrict__ init_link) {
  int t = blockIdx.x * blockDim.x + threadIdx.x;
  if (t >= NL * NS) return;
  int l = t / NS, s = t % NS;
  float at = 0.0f, ar = 0.0f;
  for (int m = 0; m < NM; ++m) {
    int fid = ftq[(l * NM + m) * 2];
    at += traffic[fid * NS + s];
    ar += rate[fid * NS + s];
  }
  float load = (at + ar * hdr[l]) / (cap[l] * 1e9f);
  float h1[NH];
  for (int j = 0; j < NH; ++j) h1[j] = reluf_(load * W1[j] + b1[j]);
  float* outp = init_link + ((size_t)s * NL + l) * NH;
  for (int j = 0; j < NH; ++j) {
    float acc = b2[j];
    for (int i = 0; i < NH; ++i) acc += h1[i] * W2[i * NH + j];
    outp[j] = reluf_(acc);
  }
}

// ---- queue embedding: one_hot(buffer_type) -> 32 -> 32, relu ----
__global__ void queue_emb_kernel(const int* __restrict__ btype,
                                 const float* __restrict__ W1, const float* __restrict__ b1,
                                 const float* __restrict__ W2, const float* __restrict__ b2,
                                 float* __restrict__ qs) {
  int q = blockIdx.x * blockDim.x + threadIdx.x;
  if (q >= NQ) return;
  int bt = btype[q];
  float h1[NH];
  for (int j = 0; j < NH; ++j) h1[j] = reluf_(W1[bt * NH + j] + b1[j]);
  for (int j = 0; j < NH; ++j) {
    float acc = b2[j];
    for (int i = 0; i < NH; ++i) acc += h1[i] * W2[i * NH + j];
    qs[(size_t)q * NH + j] = reluf_(acc);
  }
}

// ---- node embedding: sum(queue groups) -> 32 -> 32, relu ----
__global__ void node_emb_kernel(const int* __restrict__ grp, const float* __restrict__ qs,
                                const float* __restrict__ W1, const float* __restrict__ b1,
                                const float* __restrict__ W2, const float* __restrict__ b2,
                                float* __restrict__ ns) {
  int n = blockIdx.x * blockDim.x + threadIdx.x;
  if (n >= NN) return;
  float x[NH];
  for (int j = 0; j < NH; ++j) {
    float s = 0.0f;
    for (int g = 0; g < NG; ++g) s += qs[(size_t)grp[n * NG + g] * NH + j];
    x[j] = s;
  }
  float h1[NH];
  for (int j = 0; j < NH; ++j) {
    float acc = b1[j];
    for (int i = 0; i < NH; ++i) acc += x[i] * W1[i * NH + j];
    h1[j] = reluf_(acc);
  }
  for (int j = 0; j < NH; ++j) {
    float acc = b2[j];
    for (int i = 0; i < NH; ++i) acc += h1[i] * W2[i * NH + j];
    ns[(size_t)n * NH + j] = reluf_(acc);
  }
}

// ---- readout: per-flow occupancy MLP (32->16->16->1) over seq_full[:,1:], /cap, sum over P ----
__global__ void __launch_bounds__(256)
readout_kernel(const float* __restrict__ seqf, const int* __restrict__ ltp,
               const float* __restrict__ cap,
               const float* __restrict__ W1, const float* __restrict__ b1,
               const float* __restrict__ W2, const float* __restrict__ b2,
               const float* __restrict__ W3, const float* __restrict__ b3,
               float* __restrict__ out) {
  __shared__ float sW1[NH * 16];
  __shared__ float sW2[16 * 16];
  __shared__ float sW3[16];
  __shared__ float sb1[16];
  __shared__ float sb2[16];
  __shared__ float sb3;
  for (int i = threadIdx.x; i < NH * 16; i += blockDim.x) sW1[i] = W1[i];
  for (int i = threadIdx.x; i < 16 * 16; i += blockDim.x) sW2[i] = W2[i];
  if (threadIdx.x < 16) {
    sW3[threadIdx.x] = W3[threadIdx.x];
    sb1[threadIdx.x] = b1[threadIdx.x];
    sb2[threadIdx.x] = b2[threadIdx.x];
  }
  if (threadIdx.x == 0) sb3 = b3[0];
  __syncthreads();
  int f = blockIdx.x * blockDim.x + threadIdx.x;
  if (f >= NF) return;
  float d = 0.0f;
  for (int p = 0; p < NP; ++p) {
    const float* x = seqf + ((size_t)f * (NP + 1) + p + 1) * NH;
    float h1[16];
    for (int j = 0; j < 16; ++j) {
      float a = sb1[j];
      for (int i = 0; i < NH; ++i) a += x[i] * sW1[i * 16 + j];
      h1[j] = reluf_(a);
    }
    float h2[16];
    for (int j = 0; j < 16; ++j) {
      float a = sb2[j];
      for (int i = 0; i < 16; ++i) a += h1[i] * sW2[i * 16 + j];
      h2[j] = reluf_(a);
    }
    float occ = sb3;
    for (int i = 0; i < 16; ++i) occ += h2[i] * sW3[i];
    int lk = ltp[f * NP + p];
    d += occ / (cap[lk] * 1e9f);
  }
  out[f] = d;  // flow_mask is all-true by construction
}

extern "C" void kernel_launch(void* const* d_in, const int* in_sizes, int n_in,
                              void* d_out, int out_size, void* d_ws, size_t ws_size,
                              hipStream_t stream) {
  (void)out_size;
  if (n_in < 49) return;
  const float* traffic = (const float*)d_in[0];
  const float* rate = (const float*)d_in[1];
  const float* cap = (const float*)d_in[3];
  const float* hdr = (const float*)d_in[4];

  const float* P[38];
  for (int i = 0; i < 38; ++i) P[i] = (const float*)d_in[5 + i];
  // insertion-order flatten: sizes at [5..8] = 96,32,1024,32 ; jax-sorted: 96,1024,32,32
  bool sorted = (in_sizes[6] == 1024);

  const float *fe_W1, *fe_b1, *fe_W2, *fe_b2, *le_W1, *le_b1, *le_W2, *le_b2;
  const float *qe_W1, *qe_b1, *qe_W2, *qe_b2, *ne_W1, *ne_b1, *ne_W2, *ne_b2;
  const float *ro_W1, *ro_b1, *ro_W2, *ro_b2, *ro_W3, *ro_b3;
  const float *fg_Wx, *fg_Wh, *fg_bi, *fg_bh, *qg_Wx, *qg_Wh, *qg_bi, *qg_bh;
  const float *lg_Wx, *lg_Wh, *lg_bi, *lg_bh, *ng_Wx, *ng_Wh, *ng_bi, *ng_bh;
  if (!sorted) {
    fe_W1 = P[0];  fe_b1 = P[1];  fe_W2 = P[2];  fe_b2 = P[3];
    le_W1 = P[4];  le_b1 = P[5];  le_W2 = P[6];  le_b2 = P[7];
    qe_W1 = P[8];  qe_b1 = P[9];  qe_W2 = P[10]; qe_b2 = P[11];
    ne_W1 = P[12]; ne_b1 = P[13]; ne_W2 = P[14]; ne_b2 = P[15];
    ro_W1 = P[16]; ro_b1 = P[17]; ro_W2 = P[18]; ro_b2 = P[19]; ro_W3 = P[20]; ro_b3 = P[21];
    fg_Wx = P[22]; fg_Wh = P[23]; fg_bi = P[24]; fg_bh = P[25];
    qg_Wx = P[26]; qg_Wh = P[27]; qg_bi = P[28]; qg_bh = P[29];
    lg_Wx = P[30]; lg_Wh = P[31]; lg_bi = P[32]; lg_bh = P[33];
    ng_Wx = P[34]; ng_Wh = P[35]; ng_bi = P[36]; ng_bh = P[37];
  } else {
    fe_W1 = P[0];  fe_W2 = P[1];  fe_b1 = P[2];  fe_b2 = P[3];
    fg_Wh = P[4];  fg_Wx = P[5];  fg_bh = P[6];  fg_bi = P[7];
    le_W1 = P[8];  le_W2 = P[9];  le_b1 = P[10]; le_b2 = P[11];
    lg_Wh = P[12]; lg_Wx = P[13]; lg_bh = P[14]; lg_bi = P[15];
    ne_W1 = P[16]; ne_W2 = P[17]; ne_b1 = P[18]; ne_b2 = P[19];
    ng_Wh = P[20]; ng_Wx = P[21]; ng_bh = P[22]; ng_bi = P[23];
    qe_W1 = P[24]; qe_W2 = P[25]; qe_b1 = P[26]; qe_b2 = P[27];
    qg_Wh = P[28]; qg_Wx = P[29]; qg_bh = P[30]; qg_bi = P[31];
    ro_W1 = P[32]; ro_W2 = P[33]; ro_W3 = P[34]; ro_b1 = P[35]; ro_b2 = P[36]; ro_b3 = P[37];
  }

  const int* ltp = (const int*)d_in[43];     // link_to_path [F,P]
  const int* ftq = (const int*)d_in[44];     // flow_to_queue [Q,M,2]
  const int* ngrp = (const int*)d_in[45];    // node_groupings [N,G]
  const int* invng = (const int*)d_in[46];   // inverse_node_groupings [Q]
  const int* qtl = (const int*)d_in[47];     // queue_to_link [L,1]
  const int* btype = (const int*)d_in[48];   // buffer_type [Q,1]
  float* out = (float*)d_out;

  // workspace carve (128B-aligned chunks)
  float* ws = (float*)d_ws;
  size_t off = 0;
  auto take = [&](size_t n) {
    float* p = ws + off;
    off += (n + 31) & ~(size_t)31;
    return p;
  };
  float* init_flow = take((size_t)NS * NF * NH);
  float* init_link = take((size_t)NS * NL * NH);
  float* qstate = take((size_t)NQ * NH);
  float* nstate = take((size_t)NN * NH);
  float* seqf = take((size_t)NF * (NP + 1) * NH);
  float* gxall = take((size_t)NF * NP * 96);
  float* gsx = take((size_t)NQ * 96);
  float* fsum = take((size_t)NQ * NH);
  float* xn = take((size_t)NN * NH);
  unsigned int* bp_fg_wx = (unsigned int*)take(3072);
  unsigned int* bp_fg_wh = (unsigned int*)take(1536);
  unsigned int* bp_qg_wx = (unsigned int*)take(3072);
  unsigned int* bp_qg_wh = (unsigned int*)take(1536);
  unsigned int* bp_lg_wx = (unsigned int*)take(1536);
  unsigned int* bp_lg_wh = (unsigned int*)take(1536);
  unsigned int* bp_ng_wx = (unsigned int*)take(1536);
  unsigned int* bp_ng_wh = (unsigned int*)take(1536);
  if (off * sizeof(float) > ws_size) return;  // insufficient scratch

  auto cdiv = [](int a, int b) { return (a + b - 1) / b; };
  auto gwav = [&](int m) { return dim3((unsigned)((m / 16 + 7) / 8)); };  // 8 waves/block
  const int TB = 256;

  // --- pack weights into WMMA B layout (once per launch; captured in graph) ---
  pack_b_kernel<<<cdiv(3072, TB), TB, 0, stream>>>(fg_Wx, bp_fg_wx, 64);
  pack_b_kernel<<<cdiv(1536, TB), TB, 0, stream>>>(fg_Wh, bp_fg_wh, 32);
  pack_b_kernel<<<cdiv(3072, TB), TB, 0, stream>>>(qg_Wx, bp_qg_wx, 64);
  pack_b_kernel<<<cdiv(1536, TB), TB, 0, stream>>>(qg_Wh, bp_qg_wh, 32);
  pack_b_kernel<<<cdiv(1536, TB), TB, 0, stream>>>(lg_Wx, bp_lg_wx, 32);
  pack_b_kernel<<<cdiv(1536, TB), TB, 0, stream>>>(lg_Wh, bp_lg_wh, 32);
  pack_b_kernel<<<cdiv(1536, TB), TB, 0, stream>>>(ng_Wx, bp_ng_wx, 32);
  pack_b_kernel<<<cdiv(1536, TB), TB, 0, stream>>>(ng_Wh, bp_ng_wh, 32);

  // --- embeddings ---
  flow_emb_kernel<<<cdiv(NF * NS, TB), TB, 0, stream>>>(traffic, rate, fe_W1, fe_b1, fe_W2, fe_b2, init_flow);
  link_emb_kernel<<<cdiv(NL * NS, TB), TB, 0, stream>>>(ftq, traffic, rate, hdr, cap,
                                                        le_W1, le_b1, le_W2, le_b2, init_link);
  queue_emb_kernel<<<cdiv(NQ, TB), TB, 0, stream>>>(btype, qe_W1, qe_b1, qe_W2, qe_b2, qstate);
  node_emb_kernel<<<cdiv(NN, TB), TB, 0, stream>>>(ngrp, qstate, ne_W1, ne_b1, ne_W2, ne_b2, nstate);

  for (int s = 0; s < NS; ++s) {
    float* h = init_flow + (size_t)s * NF * NH;   // flow_state (mutated; recomputed each launch)
    float* ls = init_link + (size_t)s * NL * NH;  // link_state
    for (int it = 0; it < NITERS; ++it) {
      seq0_copy_kernel<<<cdiv(NF * NH, TB), TB, 0, stream>>>(h, seqf);
      // gx for all P steps in one WMMA GEMM; x = [queue_state|link_state] gathered via link_to_path
      gemm96_kernel<2><<<gwav(NF * NP), TB, 0, stream>>>(qstate, ls, ltp, ltp, 1,
                                                         bp_fg_wx, fg_bi, gxall, NF * NP);
      for (int t = 0; t < NP; ++t)
        gru_hgate_kernel<<<gwav(NF), TB, 0, stream>>>(h, bp_fg_wh, fg_bh, gxall, NP, t,
                                                      seqf + (size_t)(t + 1) * NH, (NP + 1) * NH, NF);
      // queue update
      flow_sum_kernel<<<cdiv(NQ * NH, TB), TB, 0, stream>>>(ftq, seqf, fsum);
      gemm96_kernel<2><<<gwav(NQ), TB, 0, stream>>>(fsum, nstate, nullptr, invng, 1,
                                                    bp_qg_wx, qg_bi, gsx, NQ);
      gru_hgate_kernel<<<gwav(NQ), TB, 0, stream>>>(qstate, bp_qg_wh, qg_bh, gsx, 1, 0, nullptr, 0, NQ);
      // link update (scan length 1)
      gemm96_kernel<1><<<gwav(NL), TB, 0, stream>>>(qstate, nullptr, qtl, nullptr, 1,
                                                    bp_lg_wx, lg_bi, gsx, NL);
      gru_hgate_kernel<<<gwav(NL), TB, 0, stream>>>(ls, bp_lg_wh, lg_bh, gsx, 1, 0, nullptr, 0, NL);
      // node update
      node_sum_kernel<<<cdiv(NN * NH, TB), TB, 0, stream>>>(ngrp, qstate, xn);
      gemm96_kernel<1><<<gwav(NN), TB, 0, stream>>>(xn, nullptr, nullptr, nullptr, 1,
                                                    bp_ng_wx, ng_bi, gsx, NN);
      gru_hgate_kernel<<<gwav(NN), TB, 0, stream>>>(nstate, bp_ng_wh, ng_bh, gsx, 1, 0, nullptr, 0, NN);
    }
    readout_kernel<<<cdiv(NF, TB), TB, 0, stream>>>(seqf, ltp, cap, ro_W1, ro_b1, ro_W2, ro_b2,
                                                    ro_W3, ro_b3, out + (size_t)s * NF);
  }
}
